// PipelinedMoEBlock_82145544503592
// MI455X (gfx1250) — matrix operand order repaired and checked by reading, not compiled, not run on hardware
//
#include <hip/hip_runtime.h>
#include <hip/hip_bf16.h>
#include <cstdint>

// ---------------------------------------------------------------------------
// PipelinedMoEBlock for MI455X (gfx1250): f16 WMMA + async-to-LDS staging.
// B=4 S=2048 D=1024 H=16 DH=64 E=8 DFF=4096 TOPK=2
// ---------------------------------------------------------------------------

typedef _Float16 f16;
typedef __attribute__((ext_vector_type(16))) _Float16 v16h;
typedef __attribute__((ext_vector_type(8)))  _Float16 v8h;
typedef __attribute__((ext_vector_type(8)))  float    v8f;

#define BB   4
#define SS   2048
#define DD   1024
#define HH   16
#define DHh  64
#define EE   8
#define DFFc 4096
#define TT   (BB * SS)          // 8192 tokens
#define QKVLD (3 * DD)          // 3072 halfs per qkv row

// -------------------------------- WMMA helpers -----------------------------

static __device__ __forceinline__ v8f wmma16(v16h a, v16h b, v8f c) {
  return __builtin_amdgcn_wmma_f32_16x16x32_f16(false, a, false, b, (short)0, c,
                                                false, false);
}

static __device__ __forceinline__ v16h pack16(v8h lo, v8h hi) {
  v16h a;
#pragma unroll
  for (int i = 0; i < 8; ++i) { a[i] = lo[i]; a[i + 8] = hi[i]; }
  return a;
}

// ----------------------- CDNA5 async copy primitives -----------------------
// GLOBAL_LOAD_ASYNC_TO_LDS_B128: LDS[vdst] = MEM[vaddr] (16B/lane, ASYNCcnt).
static __device__ __forceinline__ void async_b128(uint32_t lds, const f16* g) {
  asm volatile("global_load_async_to_lds_b128 %0, %1, off"
               :: "v"(lds), "v"(g) : "memory");
}
static __device__ __forceinline__ void wait_async_le4() {
  asm volatile("s_wait_asynccnt 0x4" ::: "memory");
}
static __device__ __forceinline__ void wait_async_0() {
  asm volatile("s_wait_asynccnt 0x0" ::: "memory");
}

// ----------------------- pipelined LDS-staged GEMM core --------------------
// Block: 256 threads = 8 waves (4 M-waves x 2 N-waves); block tile 128x128.
// Wave tile 32x64 -> acc[2][4]. A/B k-slices (128x32 halfs each) staged in
// LDS, double-buffered, filled by global_load_async_to_lds_b128.
// LDS map (halfs): [A buf0: 4096][A buf1: 4096][B buf0: 4096][B buf1: 4096]
//
// Fragment layouts used here (contiguous K, same order for A and B):
//  A(16x32): lane L -> row M=L%16, K contig (L<16?0..15:16..31)
//  B(32x16) from [N,K] tile: lane L -> col N=L%16, K contig (L<16?0..15:16..31)
static __device__ __forceinline__ void gemm_pipeline(
    f16* smem,
    const f16* ga0, const f16* ga1, const f16* gb0, const f16* gb1,
    uint32_t la0, uint32_t la1, uint32_t lb0, uint32_t lb1,
    int K, int lane, int wm, int wn, v8f acc[2][4]) {
  const uint32_t base = (uint32_t)(uintptr_t)smem;   // LDS byte offset
  const int l15 = lane & 15, lh = lane >> 4;
  const int nst = K / 32;
  // prologue: issue stages 0 and 1
#pragma unroll
  for (int s = 0; s < 2; ++s) {
    uint32_t ab = base + s * 8192;
    uint32_t bb = base + 16384 + s * 8192;
    async_b128(ab + la0, ga0 + s * 32);
    async_b128(ab + la1, ga1 + s * 32);
    async_b128(bb + lb0, gb0 + s * 32);
    async_b128(bb + lb1, gb1 + s * 32);
  }
  wait_async_le4();        // stage 0 landed (stage 1 still in flight)
  __syncthreads();
  for (int s = 0; s < nst; ++s) {
    const int buf = s & 1;
    const f16* aBuf = smem + buf * 4096;
    const f16* bBuf = smem + 8192 + buf * 4096;
    v16h bfr[4];
#pragma unroll
    for (int t = 0; t < 4; ++t)
      bfr[t] = *(const v16h*)(bBuf + (wn * 64 + t * 16 + l15) * 32 +
                              (lh ? 16 : 0));
#pragma unroll
    for (int sub = 0; sub < 2; ++sub) {
      const f16* ap = aBuf + (wm * 32 + sub * 16 + l15) * 32 + (lh ? 16 : 0);
      v16h a = *(const v16h*)ap;
#pragma unroll
      for (int t = 0; t < 4; ++t) acc[sub][t] = wmma16(a, bfr[t], acc[sub][t]);
    }
    __syncthreads();                 // all waves done reading buf
    if (s + 2 < nst) {
      uint32_t ab = base + buf * 8192;
      uint32_t bb = base + 16384 + buf * 8192;
      const int k0 = (s + 2) * 32;
      async_b128(ab + la0, ga0 + k0);
      async_b128(ab + la1, ga1 + k0);
      async_b128(bb + lb0, gb0 + k0);
      async_b128(bb + lb1, gb1 + k0);
      wait_async_le4();              // stage s+1 landed
    } else {
      wait_async_0();                // drain tail
    }
    __syncthreads();                 // stage s+1 visible block-wide
  }
}

// Per-thread copy-chunk descriptors: 128x32-half tile = 512 16B-chunks,
// 2 per thread. chunk c: row=c>>2, part=c&3 (part = 8-half column group).
#define CHUNKS(tid, r0, p0, r1, p1)                                           \
  const int c0_ = (tid) * 2, c1_ = c0_ + 1;                                   \
  const int r0 = c0_ >> 2, p0 = c0_ & 3, r1 = c1_ >> 2, p1 = c1_ & 3;

static __device__ __forceinline__ float gelu_tanh(float x) {
  float x3 = x * x * x;
  return 0.5f * x * (1.0f + tanhf(0.7978845608028654f * (x + 0.044715f * x3)));
}

// ------------------------- weight convert + transpose ----------------------
__global__ void convT_kernel(const float* __restrict__ src, f16* __restrict__ dst,
                             int K, int N) {
  size_t base = (size_t)blockIdx.z * (size_t)K * (size_t)N;
  long idx = (long)blockIdx.x * 256 + threadIdx.x;
  long total = (long)K * (long)N;
  if (idx >= total) return;
  int n = (int)(idx / K);
  int k = (int)(idx % K);
  dst[base + (size_t)n * K + k] = (f16)src[base + (size_t)k * N + n];
}

// ------------------------------- layernorm ---------------------------------
__global__ void ln_kernel(const float* __restrict__ xin,
                          const float* __restrict__ gg,
                          const float* __restrict__ bb,
                          f16* __restrict__ outp) {
  const int t = blockIdx.x;
  const int tid = threadIdx.x;
  const float* row = xin + (size_t)t * DD;
  __shared__ float red[256];
  __shared__ float red2[256];
  float v[4]; float s = 0.f, s2 = 0.f;
#pragma unroll
  for (int i = 0; i < 4; ++i) {
    v[i] = row[tid * 4 + i];
    s += v[i]; s2 += v[i] * v[i];
  }
  red[tid] = s; red2[tid] = s2;
  __syncthreads();
  for (int st = 128; st > 0; st >>= 1) {
    if (tid < st) { red[tid] += red[tid + st]; red2[tid] += red2[tid + st]; }
    __syncthreads();
  }
  float mu = red[0] * (1.0f / DD);
  float var = red2[0] * (1.0f / DD) - mu * mu;
  float r = rsqrtf(var + 1e-5f);
#pragma unroll
  for (int i = 0; i < 4; ++i) {
    int d = tid * 4 + i;
    outp[(size_t)t * DD + d] = (f16)((v[i] - mu) * r * gg[d] + bb[d]);
  }
}

// -------------------------- dense GEMM kernels -----------------------------
// C(f16) = A(f16 [M,K]) @ B^T(f16 [N,K]) + bias
__global__ __launch_bounds__(256) void gemm_f16out_kernel(
    const f16* __restrict__ A, const f16* __restrict__ BT,
    const float* __restrict__ bias, f16* __restrict__ C, int K, int N) {
  __shared__ __attribute__((aligned(128))) f16 smem[16384];
  const int tid = threadIdx.x;
  const int lane = tid & 31, w = tid >> 5;
  const int l15 = lane & 15, lh = lane >> 4;
  const int wm = w >> 1, wn = w & 1;
  const int mB = blockIdx.x * 128, nB = blockIdx.y * 128;
  CHUNKS(tid, r0, p0, r1, p1)
  const f16* ga0 = A + (size_t)(mB + r0) * K + p0 * 8;
  const f16* ga1 = A + (size_t)(mB + r1) * K + p1 * 8;
  const f16* gb0 = BT + (size_t)(nB + r0) * K + p0 * 8;
  const f16* gb1 = BT + (size_t)(nB + r1) * K + p1 * 8;
  const uint32_t la0 = r0 * 64 + p0 * 16, la1 = r1 * 64 + p1 * 16;
  v8f acc[2][4] = {};
  gemm_pipeline(smem, ga0, ga1, gb0, gb1, la0, la1, la0, la1, K, lane, wm, wn,
                acc);
#pragma unroll
  for (int t = 0; t < 4; ++t) {
    int n = nB + wn * 64 + t * 16 + l15;
    float bv = bias[n];
#pragma unroll
    for (int sub = 0; sub < 2; ++sub)
#pragma unroll
      for (int j = 0; j < 8; ++j) {
        int m = mB + wm * 32 + sub * 16 + j + 8 * lh;
        C[(size_t)m * N + n] = (f16)(acc[sub][t][j] + bv);
      }
  }
}

// C(f32) = resid + A @ B^T + bias
__global__ __launch_bounds__(256) void gemm_resid_kernel(
    const f16* __restrict__ A, const f16* __restrict__ BT,
    const float* __restrict__ bias, const float* __restrict__ resid,
    float* __restrict__ C, int K, int N) {
  __shared__ __attribute__((aligned(128))) f16 smem[16384];
  const int tid = threadIdx.x;
  const int lane = tid & 31, w = tid >> 5;
  const int l15 = lane & 15, lh = lane >> 4;
  const int wm = w >> 1, wn = w & 1;
  const int mB = blockIdx.x * 128, nB = blockIdx.y * 128;
  CHUNKS(tid, r0, p0, r1, p1)
  const f16* ga0 = A + (size_t)(mB + r0) * K + p0 * 8;
  const f16* ga1 = A + (size_t)(mB + r1) * K + p1 * 8;
  const f16* gb0 = BT + (size_t)(nB + r0) * K + p0 * 8;
  const f16* gb1 = BT + (size_t)(nB + r1) * K + p1 * 8;
  const uint32_t la0 = r0 * 64 + p0 * 16, la1 = r1 * 64 + p1 * 16;
  v8f acc[2][4] = {};
  gemm_pipeline(smem, ga0, ga1, gb0, gb1, la0, la1, la0, la1, K, lane, wm, wn,
                acc);
#pragma unroll
  for (int t = 0; t < 4; ++t) {
    int n = nB + wn * 64 + t * 16 + l15;
    float bv = bias[n];
#pragma unroll
    for (int sub = 0; sub < 2; ++sub)
#pragma unroll
      for (int j = 0; j < 8; ++j) {
        int m = mB + wm * 32 + sub * 16 + j + 8 * lh;
        size_t o = (size_t)m * N + n;
        C[o] = resid[o] + acc[sub][t][j] + bv;
      }
  }
}

// ------------------------------- attention ---------------------------------
__global__ __launch_bounds__(32) void attn_kernel(const f16* __restrict__ qkv,
                                                  f16* __restrict__ obuf) {
  const int qb = blockIdx.x;
  const int h  = blockIdx.y;
  const int b  = blockIdx.z;
  const int lane = threadIdx.x;
  const int l15 = lane & 15, lh = lane >> 4;
  const size_t baseTok = (size_t)b * SS;

  __shared__ __attribute__((aligned(64))) f16 probs[16 * 32];
  __shared__ __attribute__((aligned(64))) f16 vt[DHh * 32];

  const f16* qbase =
      qkv + (baseTok + qb * 16 + l15) * QKVLD + h * DHh + (lh ? 16 : 0);
  v16h aq[2];
#pragma unroll
  for (int ch = 0; ch < 2; ++ch)
    aq[ch] = *(const v16h*)(qbase + ch * 32);

  v8f o[4] = {};
  float runm[8], runl[8];
#pragma unroll
  for (int j = 0; j < 8; ++j) { runm[j] = -3.0e38f; runl[j] = 0.f; }

  for (int kb = 0; kb < SS; kb += 32) {
    v8f c0 = {}, c1 = {};
#pragma unroll
    for (int ch = 0; ch < 2; ++ch) {
      const f16* kp0 = qkv + (baseTok + kb + l15) * QKVLD + DD + h * DHh +
                       ch * 32 + (lh ? 16 : 0);
      const f16* kp1 = kp0 + 16 * QKVLD;
      c0 = wmma16(aq[ch], *(const v16h*)kp0, c0);
      c1 = wmma16(aq[ch], *(const v16h*)kp1, c1);
    }
#pragma unroll
    for (int j = 0; j < 8; ++j) { c0[j] *= 0.125f; c1[j] *= 0.125f; }

    float mx[8];
#pragma unroll
    for (int j = 0; j < 8; ++j) mx[j] = fmaxf(c0[j], c1[j]);
#pragma unroll
    for (int msk = 1; msk <= 8; msk <<= 1)
#pragma unroll
      for (int j = 0; j < 8; ++j) mx[j] = fmaxf(mx[j], __shfl_xor(mx[j], msk, 32));

    float alpha[8], p0[8], p1[8], bs[8];
#pragma unroll
    for (int j = 0; j < 8; ++j) {
      float nm = fmaxf(runm[j], mx[j]);
      alpha[j] = __expf(runm[j] - nm);
      runm[j] = nm;
      p0[j] = __expf(c0[j] - nm);
      p1[j] = __expf(c1[j] - nm);
      bs[j] = p0[j] + p1[j];
    }
#pragma unroll
    for (int msk = 1; msk <= 8; msk <<= 1)
#pragma unroll
      for (int j = 0; j < 8; ++j) bs[j] += __shfl_xor(bs[j], msk, 32);
#pragma unroll
    for (int j = 0; j < 8; ++j) runl[j] = runl[j] * alpha[j] + bs[j];
#pragma unroll
    for (int t = 0; t < 4; ++t)
#pragma unroll
      for (int j = 0; j < 8; ++j) o[t][j] *= alpha[j];

#pragma unroll
    for (int j = 0; j < 8; ++j) {
      int r = j + 8 * lh;
      probs[r * 32 + l15]      = (f16)p0[j];
      probs[r * 32 + 16 + l15] = (f16)p1[j];
    }
    {
      const f16* vp = qkv + (baseTok + kb + lane) * QKVLD + 2 * DD + h * DHh;
#pragma unroll
      for (int d0 = 0; d0 < DHh; d0 += 8) {
        v8h vv = *(const v8h*)(vp + d0);
#pragma unroll
        for (int i = 0; i < 8; ++i) vt[(d0 + i) * 32 + lane] = vv[i];
      }
    }
    __syncthreads();

    const f16* pr = probs + l15 * 32 + (lh ? 16 : 0);
    v16h ap = *(const v16h*)pr;
#pragma unroll
    for (int t = 0; t < 4; ++t) {
      const f16* vb = vt + (t * 16 + l15) * 32 + (lh ? 16 : 0);
      o[t] = wmma16(ap, *(const v16h*)vb, o[t]);
    }
    __syncthreads();
  }

#pragma unroll
  for (int j = 0; j < 8; ++j) {
    float inv = 1.0f / runl[j];
    int m = qb * 16 + j + 8 * lh;
    size_t ro = (baseTok + m) * DD + h * DHh;
#pragma unroll
    for (int t = 0; t < 4; ++t)
      obuf[ro + t * 16 + l15] = (f16)(o[t][j] * inv);
  }
}

// ------------------------------ MoE routing --------------------------------
__global__ __launch_bounds__(32) void gate_kernel(
    const f16* __restrict__ moe_in, const float* __restrict__ wg,
    int* __restrict__ tope, float* __restrict__ topw, int* __restrict__ cnt) {
  const int t = blockIdx.x;
  const int lane = threadIdx.x;
  float acc[EE];
#pragma unroll
  for (int e = 0; e < EE; ++e) acc[e] = 0.f;
  const f16* xr = moe_in + (size_t)t * DD;
  for (int d = lane; d < DD; d += 32) {
    float xv = (float)xr[d];
    const float* wr = wg + (size_t)d * EE;
#pragma unroll
    for (int e = 0; e < EE; ++e) acc[e] += xv * wr[e];
  }
#pragma unroll
  for (int msk = 16; msk >= 1; msk >>= 1)
#pragma unroll
    for (int e = 0; e < EE; ++e) acc[e] += __shfl_xor(acc[e], msk, 32);
  if (lane == 0) {
    int e0 = 0;
#pragma unroll
    for (int e = 1; e < EE; ++e) if (acc[e] > acc[e0]) e0 = e;
    int e1 = (e0 == 0) ? 1 : 0;
#pragma unroll
    for (int e = 0; e < EE; ++e)
      if (e != e0 && acc[e] > acc[e1]) e1 = e;
    float z1 = __expf(acc[e1] - acc[e0]);
    float inv = 1.0f / (1.0f + z1);
    tope[t * 2] = e0; tope[t * 2 + 1] = e1;
    topw[t * 2] = inv; topw[t * 2 + 1] = z1 * inv;
    atomicAdd(&cnt[e0], 1);
    atomicAdd(&cnt[e1], 1);
  }
}

__global__ void prefix_kernel(const int* __restrict__ cnt, int* __restrict__ off) {
  if (threadIdx.x == 0 && blockIdx.x == 0) {
    int s = 0;
    for (int e = 0; e < EE; ++e) { off[e] = s; s += cnt[e]; }
    off[EE] = s;
  }
}

__global__ void assign_kernel(const int* __restrict__ tope,
                              const float* __restrict__ topw,
                              const int* __restrict__ off, int* __restrict__ cnt2,
                              int* __restrict__ rowmap, float* __restrict__ wslot) {
  int t = blockIdx.x * blockDim.x + threadIdx.x;
  if (t >= TT) return;
#pragma unroll
  for (int k = 0; k < 2; ++k) {
    int e = tope[t * 2 + k];
    int pos = atomicAdd(&cnt2[e], 1);
    int s = off[e] + pos;
    rowmap[s] = t;
    wslot[s] = topw[t * 2 + k];
  }
}

__global__ void copy_kernel(const float* __restrict__ src, float* __restrict__ dst) {
  size_t i = (size_t)blockIdx.x * 256 + threadIdx.x;
  dst[i] = src[i];
}

// ------------------------- MoE expert GEMMs (routed) -----------------------
__global__ __launch_bounds__(256) void moe1_kernel(
    const f16* __restrict__ moe_in, const f16* __restrict__ w1T,
    const float* __restrict__ b1, f16* __restrict__ he,
    const int* __restrict__ cnt, const int* __restrict__ off,
    const int* __restrict__ rowmap) {
  const int e = blockIdx.z;
  const int cn = cnt[e];
  if ((int)blockIdx.x * 128 >= cn) return;
  const int of = off[e];
  __shared__ __attribute__((aligned(128))) f16 smem[16384];
  const int tid = threadIdx.x;
  const int lane = tid & 31, w = tid >> 5;
  const int l15 = lane & 15, lh = lane >> 4;
  const int wm = w >> 1, wn = w & 1;
  const int mB = blockIdx.x * 128, nB = blockIdx.y * 128;
  CHUNKS(tid, r0, p0, r1, p1)
  int ra0 = mB + r0; ra0 = (ra0 < cn) ? ra0 : (cn - 1);
  int ra1 = mB + r1; ra1 = (ra1 < cn) ? ra1 : (cn - 1);
  const f16* ga0 = moe_in + (size_t)rowmap[of + ra0] * DD + p0 * 8;
  const f16* ga1 = moe_in + (size_t)rowmap[of + ra1] * DD + p1 * 8;
  const f16* wte = w1T + (size_t)e * DFFc * DD;
  const f16* gb0 = wte + (size_t)(nB + r0) * DD + p0 * 8;
  const f16* gb1 = wte + (size_t)(nB + r1) * DD + p1 * 8;
  const uint32_t la0 = r0 * 64 + p0 * 16, la1 = r1 * 64 + p1 * 16;
  v8f acc[2][4] = {};
  gemm_pipeline(smem, ga0, ga1, gb0, gb1, la0, la1, la0, la1, DD, lane, wm, wn,
                acc);
#pragma unroll
  for (int t = 0; t < 4; ++t) {
    int n = nB + wn * 64 + t * 16 + l15;
    float bv = b1[(size_t)e * DFFc + n];
#pragma unroll
    for (int sub = 0; sub < 2; ++sub)
#pragma unroll
      for (int j = 0; j < 8; ++j) {
        int m = mB + wm * 32 + sub * 16 + j + 8 * lh;
        if (m < cn)
          he[(size_t)(of + m) * DFFc + n] = (f16)gelu_tanh(acc[sub][t][j] + bv);
      }
  }
}

__global__ __launch_bounds__(256) void moe2_kernel(
    const f16* __restrict__ he, const f16* __restrict__ w2T,
    const float* __restrict__ b2, float* __restrict__ outp,
    const int* __restrict__ cnt, const int* __restrict__ off,
    const int* __restrict__ rowmap, const float* __restrict__ wslot) {
  const int e = blockIdx.z;
  const int cn = cnt[e];
  if ((int)blockIdx.x * 128 >= cn) return;
  const int of = off[e];
  __shared__ __attribute__((aligned(128))) f16 smem[16384];
  const int tid = threadIdx.x;
  const int lane = tid & 31, w = tid >> 5;
  const int l15 = lane & 15, lh = lane >> 4;
  const int wm = w >> 1, wn = w & 1;
  const int mB = blockIdx.x * 128, nB = blockIdx.y * 128;
  CHUNKS(tid, r0, p0, r1, p1)
  int ra0 = mB + r0; ra0 = (ra0 < cn) ? ra0 : (cn - 1);
  int ra1 = mB + r1; ra1 = (ra1 < cn) ? ra1 : (cn - 1);
  const f16* ga0 = he + (size_t)(of + ra0) * DFFc + p0 * 8;
  const f16* ga1 = he + (size_t)(of + ra1) * DFFc + p1 * 8;
  const f16* wte = w2T + (size_t)e * DD * DFFc;
  const f16* gb0 = wte + (size_t)(nB + r0) * DFFc + p0 * 8;
  const f16* gb1 = wte + (size_t)(nB + r1) * DFFc + p1 * 8;
  const uint32_t la0 = r0 * 64 + p0 * 16, la1 = r1 * 64 + p1 * 16;
  v8f acc[2][4] = {};
  gemm_pipeline(smem, ga0, ga1, gb0, gb1, la0, la1, la0, la1, DFFc, lane, wm,
                wn, acc);
#pragma unroll
  for (int t = 0; t < 4; ++t) {
    int n = nB + wn * 64 + t * 16 + l15;
    float bv = b2[(size_t)e * DD + n];
#pragma unroll
    for (int sub = 0; sub < 2; ++sub)
#pragma unroll
      for (int j = 0; j < 8; ++j) {
        int m = mB + wm * 32 + sub * 16 + j + 8 * lh;
        if (m < cn) {
          int s = of + m;
          int tok = rowmap[s];
          float wt = wslot[s];
          atomicAdd(&outp[(size_t)tok * DD + n], wt * (acc[sub][t][j] + bv));
        }
      }
  }
}

// --------------------------------- launch ----------------------------------
extern "C" void kernel_launch(void* const* d_in, const int* in_sizes, int n_in,
                              void* d_out, int out_size, void* d_ws,
                              size_t ws_size, hipStream_t stream) {
  const float* x      = (const float*)d_in[0];
  const float* ln1_g  = (const float*)d_in[1];
  const float* ln1_b  = (const float*)d_in[2];
  const float* ln2_g  = (const float*)d_in[3];
  const float* ln2_b  = (const float*)d_in[4];
  const float* w_qkv  = (const float*)d_in[5];
  const float* b_qkv  = (const float*)d_in[6];
  const float* w_out  = (const float*)d_in[7];
  const float* b_out  = (const float*)d_in[8];
  const float* w_gate = (const float*)d_in[9];
  const float* w1     = (const float*)d_in[10];
  const float* b1     = (const float*)d_in[11];
  const float* w2     = (const float*)d_in[12];
  const float* b2     = (const float*)d_in[13];
  float* outp = (float*)d_out;

  size_t cur = 0;
  auto alloc = [&](size_t bytes) -> char* {
    size_t r = (cur + 255) & ~(size_t)255;
    cur = r + bytes;
    return (char*)d_ws + r;
  };
  f16*   hn     = (f16*)alloc((size_t)TT * DD * 2);
  f16*   wqkvT  = (f16*)alloc((size_t)QKVLD * DD * 2);
  f16*   qkvb   = (f16*)alloc((size_t)TT * QKVLD * 2);
  f16*   obuf   = (f16*)alloc((size_t)TT * DD * 2);
  f16*   woutT  = (f16*)alloc((size_t)DD * DD * 2);
  float* x1     = (float*)alloc((size_t)TT * DD * 4);
  f16*   moein  = (f16*)alloc((size_t)TT * DD * 2);
  f16*   w1T    = (f16*)alloc((size_t)EE * DFFc * DD * 2);
  f16*   w2T    = (f16*)alloc((size_t)EE * DD * DFFc * 2);
  int*   tope   = (int*)alloc((size_t)TT * 2 * 4);
  float* topw   = (float*)alloc((size_t)TT * 2 * 4);
  int*   cnt    = (int*)alloc(256);
  int*   cnt2   = (int*)alloc(256);
  int*   off    = (int*)alloc(256);
  int*   rowmap = (int*)alloc((size_t)TT * 2 * 4);
  float* wslot  = (float*)alloc((size_t)TT * 2 * 4);
  f16*   he     = (f16*)alloc((size_t)TT * 2 * DFFc * 2);

  hipMemsetAsync(cnt, 0, 256, stream);
  hipMemsetAsync(cnt2, 0, 256, stream);

  convT_kernel<<<dim3((DD * QKVLD + 255) / 256, 1, 1), 256, 0, stream>>>(
      w_qkv, wqkvT, DD, QKVLD);
  convT_kernel<<<dim3((DD * DD + 255) / 256, 1, 1), 256, 0, stream>>>(
      w_out, woutT, DD, DD);
  convT_kernel<<<dim3((DD * DFFc + 255) / 256, 1, EE), 256, 0, stream>>>(
      w1, w1T, DD, DFFc);
  convT_kernel<<<dim3((DFFc * DD + 255) / 256, 1, EE), 256, 0, stream>>>(
      w2, w2T, DFFc, DD);

  ln_kernel<<<TT, 256, 0, stream>>>(x, ln1_g, ln1_b, hn);
  gemm_f16out_kernel<<<dim3(TT / 128, QKVLD / 128), 256, 0, stream>>>(
      hn, wqkvT, b_qkv, qkvb, DD, QKVLD);
  attn_kernel<<<dim3(SS / 16, HH, BB), 32, 0, stream>>>(qkvb, obuf);
  gemm_resid_kernel<<<dim3(TT / 128, DD / 128), 256, 0, stream>>>(
      obuf, woutT, b_out, x, x1, DD, DD);

  ln_kernel<<<TT, 256, 0, stream>>>(x1, ln2_g, ln2_b, moein);
  gate_kernel<<<TT, 32, 0, stream>>>(moein, w_gate, tope, topw, cnt);
  prefix_kernel<<<1, 1, 0, stream>>>(cnt, off);
  assign_kernel<<<(TT + 255) / 256, 256, 0, stream>>>(tope, topw, off, cnt2,
                                                      rowmap, wslot);
  copy_kernel<<<(TT * DD) / 256, 256, 0, stream>>>(x1, outp);
  moe1_kernel<<<dim3(TT / 128, DFFc / 128, EE), 256, 0, stream>>>(
      moein, w1T, b1, he, cnt, off, rowmap);
  moe2_kernel<<<dim3(TT / 128, DD / 128, EE), 256, 0, stream>>>(
      he, w2T, b2, outp, cnt, off, rowmap, wslot);
}
